// MultilayerGRU_88175678587371
// MI455X (gfx1250) — compile-verified, matrix operand order checked by reading, not run on hardware
//
#include <hip/hip_runtime.h>
#include <hip/hip_bf16.h>
#include <math.h>

// ---------------------------------------------------------------------------
// MultilayerGRU on MI455X (gfx1250): bf16 WMMA everywhere, persistent scan.
// B=32, S=1024, H=1024, O=1024, L=2
// ---------------------------------------------------------------------------

typedef __attribute__((ext_vector_type(16))) __bf16 v16bf;
typedef __attribute__((ext_vector_type(8)))  __bf16 v8bf;
typedef __attribute__((ext_vector_type(8)))  float  v8f;

#define B_ 32
#define S_ 1024
#define H_ 1024
#define O_ 1024
#define L_ 2
#define NBLK 32   // persistent scan blocks (co-resident, grid barrier)

// ---- WMMA helpers ---------------------------------------------------------

__device__ __forceinline__ v8f wmma_bf16(v16bf a, v16bf b, v8f c) {
  // v_wmma_f32_16x16x32_bf16: D = A(16x32) * B(32x16) + C(16x16 f32)
  return __builtin_amdgcn_wmma_f32_16x16x32_bf16(false, a, false, b,
                                                 (short)0, c, false, false);
}

// 16-bit operand layout (ISA 7.12.2, 16x32 A / 32x16 B):
//   lanes 0-15 : row = lane,    K in {kbase+0..7, kbase+16..23}, kbase=0
//   lanes 16-31: row = lane-16, same pattern with kbase=8
// => two contiguous 8-element (16B) runs per lane.
// For B we exploit B[k][n] = W[n][k] with W row-major (N x K): same loader.
__device__ __forceinline__ v16bf load_frag_row(const __bf16* __restrict__ base,
                                               int ld, int row0, int k0, int lane) {
  const int r    = lane & 15;
  const int koff = (lane >> 4) * 8;
  const __bf16* p = base + (size_t)(row0 + r) * ld + (k0 + koff);
  v8bf lo = *(const v8bf*)p;          // K = k0+koff .. +7
  v8bf hi = *(const v8bf*)(p + 16);   // K = k0+koff+16 .. +23
  v16bf out;
#pragma unroll
  for (int i = 0; i < 8; ++i) { out[i] = lo[i]; out[i + 8] = hi[i]; }
  return out;
}

// C/D f32 16x16 layout: VGPR v, lane l -> M = v + 8*(l>>4), N = l&15
__device__ __forceinline__ void store_tile_bf16(__bf16* __restrict__ C, int ldc,
                                                int m0, int n0, int lane, v8f acc) {
  const int n  = n0 + (lane & 15);
  const int mr = m0 + (lane >> 4) * 8;
#pragma unroll
  for (int v = 0; v < 8; ++v) C[(size_t)(mr + v) * ldc + n] = (__bf16)acc[v];
}

__device__ __forceinline__ void store_tile_f32_bias(float* __restrict__ C,
                                                    const float* __restrict__ bias,
                                                    int ldc, int m0, int n0,
                                                    int lane, v8f acc) {
  const int n  = n0 + (lane & 15);
  const int mr = m0 + (lane >> 4) * 8;
  const float bv = bias[n];
#pragma unroll
  for (int v = 0; v < 8; ++v) C[(size_t)(mr + v) * ldc + n] = acc[v] + bv;
}

// ---- Utility kernels ------------------------------------------------------

__global__ void k_f32_to_bf16(const float* __restrict__ in,
                              __bf16* __restrict__ out, size_t n) {
  size_t i = (size_t)blockIdx.x * blockDim.x + threadIdx.x;
  size_t stride = (size_t)gridDim.x * blockDim.x;
  for (; i < n; i += stride) out[i] = (__bf16)in[i];
}

__global__ void k_init_h(const float* __restrict__ hidden,  // (B, L, H)
                         float* __restrict__ hf, __bf16* __restrict__ hb, int l) {
  int i = blockIdx.x * 256 + threadIdx.x;     // B_*H_ threads
  int b = i >> 10;                            // H_==1024
  int j = i & (H_ - 1);
  float v = hidden[((size_t)b * L_ + l) * H_ + j];
  hf[i] = v;
  hb[i] = (__bf16)v;
}

__global__ void k_zero_barrier(unsigned* bar) {
  if (threadIdx.x == 0) { bar[0] = 0u; bar[1] = 0u; }
}

// ---- Big GEMM: C(MxN) = A(MxK,bf16) * W(NxK,bf16)^T ----------------------
// 256 threads = 8 waves in 4(M) x 2(N); wave tile 32x32 (4 accums).
// Block tile 128x64. grid = (N/64, M/128).

__global__ void __launch_bounds__(256) k_gemm_bf16(const __bf16* __restrict__ A,
                                                   const __bf16* __restrict__ W,
                                                   __bf16* __restrict__ C,
                                                   int M, int N, int K) {
  const int lane = threadIdx.x & 31;
  const int wave = threadIdx.x >> 5;
  const int m0 = blockIdx.y * 128 + (wave >> 1) * 32;
  const int n0 = blockIdx.x * 64  + (wave & 1) * 32;
  v8f c00 = {}, c01 = {}, c10 = {}, c11 = {};
  for (int k0 = 0; k0 < K; k0 += 32) {
    v16bf a0 = load_frag_row(A, K, m0,      k0, lane);
    v16bf a1 = load_frag_row(A, K, m0 + 16, k0, lane);
    v16bf b0 = load_frag_row(W, K, n0,      k0, lane);
    v16bf b1 = load_frag_row(W, K, n0 + 16, k0, lane);
    c00 = wmma_bf16(a0, b0, c00);
    c01 = wmma_bf16(a0, b1, c01);
    c10 = wmma_bf16(a1, b0, c10);
    c11 = wmma_bf16(a1, b1, c11);
  }
  store_tile_bf16(C, N, m0,      n0,      lane, c00);
  store_tile_bf16(C, N, m0,      n0 + 16, lane, c01);
  store_tile_bf16(C, N, m0 + 16, n0,      lane, c10);
  store_tile_bf16(C, N, m0 + 16, n0 + 16, lane, c11);
}

__global__ void __launch_bounds__(256) k_gemm_f32_bias(const __bf16* __restrict__ A,
                                                       const __bf16* __restrict__ W,
                                                       const float* __restrict__ bias,
                                                       float* __restrict__ C,
                                                       int M, int N, int K) {
  const int lane = threadIdx.x & 31;
  const int wave = threadIdx.x >> 5;
  const int m0 = blockIdx.y * 128 + (wave >> 1) * 32;
  const int n0 = blockIdx.x * 64  + (wave & 1) * 32;
  v8f c00 = {}, c01 = {}, c10 = {}, c11 = {};
  for (int k0 = 0; k0 < K; k0 += 32) {
    v16bf a0 = load_frag_row(A, K, m0,      k0, lane);
    v16bf a1 = load_frag_row(A, K, m0 + 16, k0, lane);
    v16bf b0 = load_frag_row(W, K, n0,      k0, lane);
    v16bf b1 = load_frag_row(W, K, n0 + 16, k0, lane);
    c00 = wmma_bf16(a0, b0, c00);
    c01 = wmma_bf16(a0, b1, c01);
    c10 = wmma_bf16(a1, b0, c10);
    c11 = wmma_bf16(a1, b1, c11);
  }
  store_tile_f32_bias(C, bias, N, m0,      n0,      lane, c00);
  store_tile_f32_bias(C, bias, N, m0,      n0 + 16, lane, c01);
  store_tile_f32_bias(C, bias, N, m0 + 16, n0,      lane, c10);
  store_tile_f32_bias(C, bias, N, m0 + 16, n0 + 16, lane, c11);
}

// ---- Grid-wide barrier (persistent scan kernel) ---------------------------

__device__ __forceinline__ void grid_barrier(unsigned* cnt, unsigned* gen) {
  __syncthreads();
  if (threadIdx.x == 0) {
    __threadfence();
    unsigned g = __hip_atomic_load(gen, __ATOMIC_ACQUIRE, __HIP_MEMORY_SCOPE_AGENT);
    unsigned a = __hip_atomic_fetch_add(cnt, 1u, __ATOMIC_ACQ_REL, __HIP_MEMORY_SCOPE_AGENT);
    if (a == NBLK - 1) {
      __hip_atomic_store(cnt, 0u, __ATOMIC_RELAXED, __HIP_MEMORY_SCOPE_AGENT);
      __hip_atomic_fetch_add(gen, 1u, __ATOMIC_ACQ_REL, __HIP_MEMORY_SCOPE_AGENT);
    } else {
      while (__hip_atomic_load(gen, __ATOMIC_ACQUIRE, __HIP_MEMORY_SCOPE_AGENT) == g)
        __builtin_amdgcn_s_sleep(1);
    }
  }
  __syncthreads();
}

// ---- Persistent GRU scan --------------------------------------------------
// 32 blocks x 8 waves = 256 waves.
// Phase A (256 units): gate in {z,r}, 64 N-tiles x 2 M-tiles each; K=1024.
//   z -> z_f32 ; r -> hr_bf = bf16(sigmoid(r_pre) * h).
// Phase B (128 units): g = tanh(xg + hr@Whg^T + bg); h' = z*h + (1-z)*g.
// Recurrent weights (3 x 2MB bf16) stay L2-resident across all 1024 steps.

__global__ void __launch_bounds__(256) k_gru_scan(
    const __bf16* __restrict__ XZ, const __bf16* __restrict__ XR,
    const __bf16* __restrict__ XG,
    const __bf16* __restrict__ Whz, const __bf16* __restrict__ Whr,
    const __bf16* __restrict__ Whg,
    const float* __restrict__ bz, const float* __restrict__ br,
    const float* __restrict__ bg,
    float* __restrict__ h_f32, __bf16* __restrict__ h_bf,
    __bf16* __restrict__ hr_bf, float* __restrict__ z_f32,
    __bf16* __restrict__ Xout, float* __restrict__ finals,
    unsigned* __restrict__ bar) {
  const int lane = threadIdx.x & 31;          // wave32
  const int wave = threadIdx.x >> 5;
  const int g = blockIdx.x * 8 + wave;        // global wave id: 0..255
  unsigned* cnt = bar;
  unsigned* gen = bar + 1;

  for (int t = 0; t < S_; ++t) {
    // ---------------- Phase A: z and r ----------------
    {
      const int gate = g >> 7;                // 0 = z, 1 = r
      const int rem  = g & 127;
      const int n0 = (rem >> 1) * 16;
      const int m0 = (rem & 1) * 16;
      const __bf16* W  = gate ? Whr : Whz;
      const __bf16* X  = gate ? XR  : XZ;
      const float*  bb = gate ? br  : bz;
      v8f acc = {};
      for (int k0 = 0; k0 < H_; k0 += 32) {
        v16bf a = load_frag_row(h_bf, H_, m0, k0, lane);
        v16bf b = load_frag_row(W,    H_, n0, k0, lane);
        acc = wmma_bf16(a, b, acc);
      }
      const int n   = n0 + (lane & 15);
      const int mrw = m0 + (lane >> 4) * 8;
      const float bias = bb[n];
#pragma unroll
      for (int v = 0; v < 8; ++v) {
        const int m = mrw + v;                // batch index 0..31
        float pre = acc[v] + (float)X[((size_t)m * S_ + t) * H_ + n] + bias;
        float s = 1.0f / (1.0f + __expf(-pre));
        if (gate == 0) {
          z_f32[m * H_ + n] = s;
        } else {
          hr_bf[m * H_ + n] = (__bf16)(s * h_f32[m * H_ + n]);
        }
      }
    }
    grid_barrier(cnt, gen);

    // ---------------- Phase B: g gate + state update ----------------
    if (g < 128) {
      const int n0 = (g >> 1) * 16;
      const int m0 = (g & 1) * 16;
      v8f acc = {};
      for (int k0 = 0; k0 < H_; k0 += 32) {
        v16bf a = load_frag_row(hr_bf, H_, m0, k0, lane);
        v16bf b = load_frag_row(Whg,   H_, n0, k0, lane);
        acc = wmma_bf16(a, b, acc);
      }
      const int n   = n0 + (lane & 15);
      const int mrw = m0 + (lane >> 4) * 8;
      const float bias = bg[n];
#pragma unroll
      for (int v = 0; v < 8; ++v) {
        const int m = mrw + v;
        float gv = tanhf(acc[v] + (float)XG[((size_t)m * S_ + t) * H_ + n] + bias);
        float zv = z_f32[m * H_ + n];
        float hv = h_f32[m * H_ + n];
        float hn = zv * hv + (1.0f - zv) * gv;
        h_f32[m * H_ + n] = hn;
        h_bf[m * H_ + n]  = (__bf16)hn;
        Xout[((size_t)m * S_ + t) * H_ + n] = (__bf16)hn;
        if (t == S_ - 1) finals[(size_t)m * (L_ * H_) + n] = hn;
      }
    }
    grid_barrier(cnt, gen);
  }
}

// ---------------------------------------------------------------------------

extern "C" void kernel_launch(void* const* d_in, const int* in_sizes, int n_in,
                              void* d_out, int out_size, void* d_ws, size_t ws_size,
                              hipStream_t stream) {
  const float* x      = (const float*)d_in[0];   // (B,S,H)
  const float* hidden = (const float*)d_in[1];   // (B,L,H)
  const float* Wxz    = (const float*)d_in[2];   // (L,H,H)
  const float* Whz    = (const float*)d_in[3];
  const float* bz     = (const float*)d_in[4];   // (L,H)
  const float* Wxr    = (const float*)d_in[5];
  const float* Whr    = (const float*)d_in[6];
  const float* br     = (const float*)d_in[7];
  const float* Wxg    = (const float*)d_in[8];
  const float* Whg    = (const float*)d_in[9];
  const float* bg     = (const float*)d_in[10];
  const float* Why    = (const float*)d_in[11];  // (O,H)
  const float* by     = (const float*)d_in[12];  // (O,)
  float* out = (float*)d_out;                    // out (B,S,O) ++ finals (B,L,H)

  // ---- workspace layout ----
  char* ws = (char*)d_ws;
  size_t off = 0;
  auto alloc = [&](size_t bytes) -> char* {
    char* p = ws + off;
    off = (off + bytes + 255) & ~(size_t)255;
    return p;
  };
  unsigned* bar   = (unsigned*)alloc(256);
  float*  h_f32   = (float*)alloc((size_t)B_ * H_ * 4);
  float*  z_f32   = (float*)alloc((size_t)B_ * H_ * 4);
  __bf16* h_bf    = (__bf16*)alloc((size_t)B_ * H_ * 2);
  __bf16* hr_bf   = (__bf16*)alloc((size_t)B_ * H_ * 2);
  __bf16* wbf[13];
  for (int i = 0; i < 13; ++i) wbf[i] = (__bf16*)alloc((size_t)H_ * H_ * 2);
  __bf16* X0  = (__bf16*)alloc((size_t)B_ * S_ * H_ * 2);
  __bf16* X1  = (__bf16*)alloc((size_t)B_ * S_ * H_ * 2);
  __bf16* XZb = (__bf16*)alloc((size_t)B_ * S_ * H_ * 2);
  __bf16* XRb = (__bf16*)alloc((size_t)B_ * S_ * H_ * 2);
  __bf16* XGb = (__bf16*)alloc((size_t)B_ * S_ * H_ * 2);

  const size_t HH = (size_t)H_ * H_;
  const int M  = B_ * S_;

  auto conv = [&](const float* src, __bf16* dst, size_t n) {
    k_f32_to_bf16<<<4096, 256, 0, stream>>>(src, dst, n);
  };

  k_zero_barrier<<<1, 64, 0, stream>>>(bar);

  // bf16 weight copies: wbf[l*6 + {0..5}] = {Wxz,Wxr,Wxg,Whz,Whr,Whg}[l], wbf[12]=Why
  for (int l = 0; l < L_; ++l) {
    conv(Wxz + l * HH, wbf[l * 6 + 0], HH);
    conv(Wxr + l * HH, wbf[l * 6 + 1], HH);
    conv(Wxg + l * HH, wbf[l * 6 + 2], HH);
    conv(Whz + l * HH, wbf[l * 6 + 3], HH);
    conv(Whr + l * HH, wbf[l * 6 + 4], HH);
    conv(Whg + l * HH, wbf[l * 6 + 5], HH);
  }
  conv(Why, wbf[12], HH);
  conv(x, X0, (size_t)B_ * S_ * H_);

  dim3 gemm_block(256);
  dim3 gemm_grid(H_ / 64, M / 128);   // (16, 256)

  for (int l = 0; l < L_; ++l) {
    const __bf16* Xin  = (l == 0) ? X0 : X1;
    __bf16*       Xout = (l == 0) ? X1 : X0;

    // Sequence-parallel input projections: xz/xr/xg = Xin @ Wx*^T
    k_gemm_bf16<<<gemm_grid, gemm_block, 0, stream>>>(Xin, wbf[l * 6 + 0], XZb, M, H_, H_);
    k_gemm_bf16<<<gemm_grid, gemm_block, 0, stream>>>(Xin, wbf[l * 6 + 1], XRb, M, H_, H_);
    k_gemm_bf16<<<gemm_grid, gemm_block, 0, stream>>>(Xin, wbf[l * 6 + 2], XGb, M, H_, H_);

    k_init_h<<<(B_ * H_) / 256, 256, 0, stream>>>(hidden, h_f32, h_bf, l);

    float* finals = out + (size_t)B_ * S_ * O_ + (size_t)l * H_;  // stride L_*H_ per batch
    k_gru_scan<<<NBLK, 256, 0, stream>>>(XZb, XRb, XGb,
                                         wbf[l * 6 + 3], wbf[l * 6 + 4], wbf[l * 6 + 5],
                                         bz + l * H_, br + l * H_, bg + l * H_,
                                         h_f32, h_bf, hr_bf, z_f32,
                                         Xout, finals, bar);
  }

  // out = hs_layer1 @ Why^T + by  (hs of layer 1 lives in X0 after the ping-pong)
  dim3 out_grid(O_ / 64, M / 128);
  k_gemm_f32_bias<<<out_grid, gemm_block, 0, stream>>>(X0, wbf[12], by, out, M, O_, H_);
}
